// GSLNet_86766929314295
// MI455X (gfx1250) — matrix operand order repaired and checked by reading, not compiled, not run on hardware
//
#include <hip/hip_runtime.h>

// Problem constants (match reference)
#define Bb 16
#define Tt 64
#define Nn 2000
#define Hh 128
#define Cc 8
#define NT 125          // Nn / 16
#define KMAX 16
#define LDSLD 2008      // 2000 + 8 pad, dodges LDS bank alignment
#define EPSF 1e-8f

typedef __attribute__((ext_vector_type(16))) __bf16 v16bf;
typedef __attribute__((ext_vector_type(8)))  __bf16 v8bf;
typedef __attribute__((ext_vector_type(8)))  float  v8f;

__device__ __forceinline__ v8f wmma_bf16(v16bf a, v16bf b, v8f c) {
  // v_wmma_f32_16x16x32_bf16 : D = A(16x32) * B(32x16) + C, fp32 accum
  return __builtin_amdgcn_wmma_f32_16x16x32_bf16(false, a, false, b, (short)0, c, false, false);
}
__device__ __forceinline__ __bf16 tobf(float f) { return (__bf16)f; }

// Fragment layouts (wave32, CDNA5 ISA):
//  A (16x32 bf16): lane row = lane&15, kbase = (lane>>4)*8; element e -> K = kbase + e + (e&8)
//    => elements 0..7 and 8..15 are two contiguous 16B runs of the source row.
//  B (32x16 bf16): lane col = lane&15, kbase = (lane>>4)*8; element e -> K = kbase + e + (e&8)
//    => two contiguous 16B runs of the (K-major) source row, same as A.
//  C/D (16x16 f32): lane col = lane&15; element r -> row = (lane>>4)*8 + r

__device__ __forceinline__ v16bf load_afrag(const __bf16* rowp, int kt, int lhi) {
  v8bf lo = *(const v8bf*)(rowp + kt * 32 + lhi * 8);
  v8bf hi = *(const v8bf*)(rowp + kt * 32 + 16 + lhi * 8);
  return __builtin_shufflevector(lo, hi, 0, 1, 2, 3, 4, 5, 6, 7,
                                 8, 9, 10, 11, 12, 13, 14, 15);
}
__device__ __forceinline__ v16bf load_bfrag(const __bf16* colrowp, int kt, int lhi) {
  v8bf lo = *(const v8bf*)(colrowp + kt * 32 + lhi * 8);
  v8bf hi = *(const v8bf*)(colrowp + kt * 32 + 16 + lhi * 8);
  return __builtin_shufflevector(lo, hi, 0, 1, 2, 3, 4, 5, 6, 7,
                                 8, 9, 10, 11, 12, 13, 14, 15);
}

// ------- one-time weight prep: bf16 + transpose (K-major for B frags) ---
__global__ void k_prep_weights(const float* __restrict__ Wenc, const float* __restrict__ Wg0,
                               const float* __restrict__ Wg1, __bf16* __restrict__ WencT,
                               __bf16* __restrict__ Wg0T, __bf16* __restrict__ Wg1T) {
  int i = blockIdx.x * 256 + threadIdx.x;
  if (i < Tt * Hh) {                       // Wenc (T,H) -> WencT (H,T)
    int t = i / Hh, h = i % Hh;
    WencT[h * Tt + t] = tobf(Wenc[i]);
  }
  if (i < Hh * Hh) {                       // Wg (H,H) -> WgT (H,H) transposed
    int k = i / Hh, h = i % Hh;
    Wg0T[h * Hh + k] = tobf(Wg0[i]);
    Wg1T[h * Hh + k] = tobf(Wg1[i]);
  }
}

// ---------------- encoder: emb = x^T @ W_enc + b_enc --------------------
__global__ void k_encode(const float* __restrict__ x, const __bf16* __restrict__ WencT,
                         const float* __restrict__ benc, float* __restrict__ emb) {
  int tile = blockIdx.x * 4 + (threadIdx.x >> 5);        // 4 waves / block
  int b  = tile / NT;
  int n0 = (tile % NT) * 16;
  int lane = threadIdx.x & 31, lrow = lane & 15, lhi = lane >> 4;

  v16bf a[2];
#pragma unroll
  for (int kt = 0; kt < 2; ++kt)
#pragma unroll
    for (int e = 0; e < 16; ++e) {
      int K = kt * 32 + lhi * 8 + e + (e & 8);
      a[kt][e] = tobf(x[((size_t)b * Tt + K) * Nn + n0 + lrow]);   // x is (B,T,N): x^T[n][t]
    }
#pragma unroll
  for (int ht = 0; ht < 8; ++ht) {
    const __bf16* wrow = WencT + (size_t)(ht * 16 + lrow) * Tt;
    v8f acc = {};
#pragma unroll
    for (int kt = 0; kt < 2; ++kt)
      acc = wmma_bf16(a[kt], load_bfrag(wrow, kt, lhi), acc);
    float bias = benc[ht * 16 + lrow];
#pragma unroll
    for (int r = 0; r < 8; ++r) {
      int node = n0 + lhi * 8 + r;
      emb[((size_t)b * Nn + node) * Hh + ht * 16 + lrow] = acc[r] + bias;
    }
  }
}

// ---------------- z = emb / (||emb|| + eps), stored bf16 ----------------
__global__ void k_normalize(const float* __restrict__ emb, __bf16* __restrict__ z) {
  int node = blockIdx.x, h = threadIdx.x;            // 128 threads
  __shared__ float red[Hh];
  float v = emb[(size_t)node * Hh + h];
  red[h] = v * v;
  __syncthreads();
  for (int s = Hh / 2; s > 0; s >>= 1) {
    if (h < s) red[h] += red[h + s];
    __syncthreads();
  }
  float nrm = sqrtf(red[0]) + EPSF;
  z[(size_t)node * Hh + h] = tobf(v / nrm);
}

// -------- S = z z^T row-stripe in LDS, top-k threshold, A + nbr lists ---
__global__ void k_sim_topk(const __bf16* __restrict__ z, const int* __restrict__ topk_p,
                           float* __restrict__ Aout, int* __restrict__ nbr_idx,
                           float* __restrict__ nbr_w) {
  extern __shared__ float smem[];
  float* Sm     = smem;                 // 16 x LDSLD
  float* kth    = Sm + 16 * LDSLD;      // 16
  float* rowsum = kth + 16;             // 16
  int*   cnt    = (int*)(rowsum + 16);  // 16

  int b  = blockIdx.x / NT;
  int n0 = (blockIdx.x % NT) * 16;
  int tid = threadIdx.x;
  int wid = tid >> 5, lane = tid & 31, lrow = lane & 15, lhi = lane >> 4;
  int K = topk_p[0];
  if (K < 1) K = 1;
  if (K > KMAX) K = KMAX;

  if (tid < 16) { cnt[tid] = 0; rowsum[tid] = 0.f; }
  for (int i = tid; i < 16 * KMAX; i += 256) {        // pre-zero neighbor slots
    size_t base = ((size_t)b * Nn + n0 + i / KMAX) * KMAX + (i % KMAX);
    nbr_idx[base] = 0;
    nbr_w[base]   = 0.f;
  }

  const __bf16* zb = z + (size_t)b * Nn * Hh;

  // A-frags: 16 rows of z, K = 128 (4 k-steps), loaded once per wave
  const __bf16* arow = zb + (size_t)(n0 + lrow) * Hh;
  v16bf a[4];
#pragma unroll
  for (int kt = 0; kt < 4; ++kt) a[kt] = load_afrag(arow, kt, lhi);

  // 125 column tiles split across the 8 waves; stage S stripe in LDS.
  // Software-pipelined: B-frags for tile ct+8 are issued before the WMMAs
  // for tile ct, hiding L2 latency (only ~2 waves/SIMD here).
  v16bf bb0[4], bb1[4];
  {
    int ct = wid;                                   // wid < 8 <= NT always
    const __bf16* brow = zb + (size_t)(ct * 16 + lrow) * Hh;
#pragma unroll
    for (int kt = 0; kt < 4; ++kt) bb0[kt] = load_bfrag(brow, kt, lhi);
  }
  for (int ct = wid; ct < NT; ct += 8) {
    int cn = ct + 8;
    if (cn < NT) {                                  // wave-uniform guard
      const __bf16* brow = zb + (size_t)(cn * 16 + lrow) * Hh;
#pragma unroll
      for (int kt = 0; kt < 4; ++kt) bb1[kt] = load_bfrag(brow, kt, lhi);
    }
    v8f acc = {};
#pragma unroll
    for (int kt = 0; kt < 4; ++kt)
      acc = wmma_bf16(a[kt], bb0[kt], acc);
#pragma unroll
    for (int r = 0; r < 8; ++r)
      Sm[(lhi * 8 + r) * LDSLD + ct * 16 + lrow] = acc[r];
#pragma unroll
    for (int kt = 0; kt < 4; ++kt) bb0[kt] = bb1[kt];
  }
  __syncthreads();

  // top-k threshold per row: wave `wid` owns rows 2*wid and 2*wid+1
  for (int rr = 0; rr < 2; ++rr) {
    int row = wid * 2 + rr;
    float loc[KMAX];
#pragma unroll
    for (int i = 0; i < KMAX; ++i) loc[i] = -1e30f;
    for (int c = lane; c < Nn; c += 32) {              // per-lane sorted top-16
      float v = Sm[row * LDSLD + c];
      if (v > loc[KMAX - 1]) {
        loc[KMAX - 1] = v;
#pragma unroll
        for (int i = KMAX - 1; i > 0; --i) {
          float hi = fmaxf(loc[i - 1], loc[i]);
          float lo = fminf(loc[i - 1], loc[i]);
          loc[i - 1] = hi; loc[i] = lo;
        }
      }
    }
    float kv = -1e30f;
    for (int t = 0; t < K; ++t) {                      // pop global max K times
      float mx = loc[0];
#pragma unroll
      for (int o = 16; o > 0; o >>= 1) mx = fmaxf(mx, __shfl_xor(mx, o, 32));
      kv = mx;
      unsigned long long bal = __ballot(loc[0] == mx);
      int first = __ffsll(bal) - 1;
      if (lane == first) {
#pragma unroll
        for (int i = 0; i < KMAX - 1; ++i) loc[i] = loc[i + 1];
        loc[KMAX - 1] = -1e30f;
      }
    }
    if (lane == 0) kth[row] = kv;
  }
  __syncthreads();

  // row sums of relu(S) over entries >= kth (16 threads per row)
  {
    int row = tid >> 4, sub = tid & 15;
    float kv = kth[row];
    float local = 0.f;
    for (int c = sub; c < Nn; c += 16) {
      float v = Sm[row * LDSLD + c];
      if (v >= kv) local += fmaxf(v, 0.f);
    }
#pragma unroll
    for (int o = 8; o > 0; o >>= 1) local += __shfl_down(local, o, 16);
    if (sub == 0) rowsum[row] = local;
  }
  __syncthreads();

  // write A (float4, coalesced) and capture sparse neighbor lists
  for (int row = 0; row < 16; ++row) {
    float kv  = kth[row];
    float inv = 1.f / (rowsum[row] + EPSF);
    size_t arowoff = ((size_t)b * Nn + n0 + row) * Nn;
    size_t nbase   = ((size_t)b * Nn + n0 + row) * KMAX;
    float4* arow4 = (float4*)(Aout + arowoff);
    for (int c4 = tid; c4 < Nn / 4; c4 += 256) {
      int c = c4 * 4;
      float4 av;
      float* avp = (float*)&av;
#pragma unroll
      for (int q = 0; q < 4; ++q) {
        float v = Sm[row * LDSLD + c + q];
        float a = (v >= kv && v > 0.f) ? v * inv : 0.f;
        avp[q] = a;
        if (a > 0.f) {
          int slot = atomicAdd(&cnt[row], 1);
          if (slot < KMAX) { nbr_idx[nbase + slot] = c + q; nbr_w[nbase + slot] = a; }
        }
      }
      arow4[c4] = av;
    }
  }
}

// ---------------- sparse A @ h via neighbor lists -----------------------
__global__ void k_gather(const float* __restrict__ hin, const int* __restrict__ nidx,
                         const float* __restrict__ nw, __bf16* __restrict__ mout) {
  int node = blockIdx.x, b = node / Nn, h = threadIdx.x;   // 128 threads
  size_t nb = (size_t)node * KMAX;
  float acc = 0.f;
#pragma unroll
  for (int s = 0; s < KMAX; ++s) {
    float wgt = nw[nb + s];
    int   j   = nidx[nb + s];
    acc += wgt * hin[((size_t)b * Nn + j) * Hh + h];
  }
  mout[(size_t)node * Hh + h] = tobf(acc);
}

// ---------------- h_out = relu(m @ W(128x128) + b) ----------------------
__global__ void k_gemm_relu(const __bf16* __restrict__ m, const __bf16* __restrict__ WT,
                            const float* __restrict__ bias, float* __restrict__ hout) {
  int tile = blockIdx.x * 4 + (threadIdx.x >> 5);        // 4 waves / block
  int b  = tile / NT;
  int n0 = (tile % NT) * 16;
  int lane = threadIdx.x & 31, lrow = lane & 15, lhi = lane >> 4;

  const __bf16* arow = m + ((size_t)b * Nn + n0 + lrow) * Hh;
  v16bf a[4];
#pragma unroll
  for (int kt = 0; kt < 4; ++kt) a[kt] = load_afrag(arow, kt, lhi);

#pragma unroll
  for (int ht = 0; ht < 8; ++ht) {
    const __bf16* wrow = WT + (size_t)(ht * 16 + lrow) * Hh;   // WT is (H_out, K)
    v8f acc = {};
#pragma unroll
    for (int kt = 0; kt < 4; ++kt)
      acc = wmma_bf16(a[kt], load_bfrag(wrow, kt, lhi), acc);
    float bv = bias[ht * 16 + lrow];
#pragma unroll
    for (int r = 0; r < 8; ++r) {
      int node = n0 + lhi * 8 + r;
      hout[((size_t)b * Nn + node) * Hh + ht * 16 + lrow] = fmaxf(acc[r] + bv, 0.f);
    }
  }
}

// ---------------- MLP head: relu(h@W1+b1)@W2+b2, transpose to (B,C,N) ---
__global__ void k_head(const float* __restrict__ h2, const float* __restrict__ W1,
                       const float* __restrict__ b1, const float* __restrict__ W2,
                       const float* __restrict__ b2, float* __restrict__ logits) {
  int node = blockIdx.x, b = node / Nn, n = node % Nn;
  __shared__ float u[Hh / 2];
  int j = threadIdx.x;                                   // 64 threads
  const float* hp = h2 + (size_t)node * Hh;
  float acc = b1[j];
  for (int h = 0; h < Hh; ++h) acc += hp[h] * W1[h * (Hh / 2) + j];
  u[j] = fmaxf(acc, 0.f);
  __syncthreads();
  if (j < Cc) {
    float l = b2[j];
#pragma unroll
    for (int t = 0; t < Hh / 2; ++t) l += u[t] * W2[t * Cc + j];
    logits[((size_t)b * Cc + j) * Nn + n] = l;
  }
}

extern "C" void kernel_launch(void* const* d_in, const int* /*in_sizes*/, int /*n_in*/,
                              void* d_out, int /*out_size*/, void* d_ws, size_t /*ws_size*/,
                              hipStream_t stream) {
  const float* x    = (const float*)d_in[0];
  const float* Wenc = (const float*)d_in[1];
  const float* benc = (const float*)d_in[2];
  const float* Wg0  = (const float*)d_in[3];
  const float* bg0  = (const float*)d_in[4];
  const float* Wg1  = (const float*)d_in[5];
  const float* bg1  = (const float*)d_in[6];
  const float* Wh1  = (const float*)d_in[7];
  const float* bh1  = (const float*)d_in[8];
  const float* Wh2  = (const float*)d_in[9];
  const float* bh2  = (const float*)d_in[10];
  const int*   topk = (const int*)d_in[11];

  float* out    = (float*)d_out;                 // (logits, A, emb) concatenated
  float* logits = out;
  float* Aout   = out + (size_t)Bb * Cc * Nn;
  float* emb    = Aout + (size_t)Bb * Nn * Nn;

  char* w = (char*)d_ws;
  auto take = [&](size_t bytes) -> char* {
    char* p = w;
    w += (bytes + 255) & ~(size_t)255;
    return p;
  };
  const size_t BNH = (size_t)Bb * Nn * Hh;
  __bf16* z     = (__bf16*)take(BNH * 2);
  int*    nidx  = (int*)take((size_t)Bb * Nn * KMAX * 4);
  float*  nw    = (float*)take((size_t)Bb * Nn * KMAX * 4);
  __bf16* mbuf  = (__bf16*)take(BNH * 2);
  float*  h1    = (float*)take(BNH * 4);
  float*  h2b   = (float*)take(BNH * 4);
  __bf16* WencT = (__bf16*)take((size_t)Tt * Hh * 2);
  __bf16* Wg0T  = (__bf16*)take((size_t)Hh * Hh * 2);
  __bf16* Wg1T  = (__bf16*)take((size_t)Hh * Hh * 2);

  k_prep_weights<<<dim3((Hh * Hh + 255) / 256), dim3(256), 0, stream>>>(
      Wenc, Wg0, Wg1, WencT, Wg0T, Wg1T);

  dim3 gTiles(Bb * NT / 4), b128(128);                 // 4 waves per block
  k_encode<<<gTiles, b128, 0, stream>>>(x, WencT, benc, emb);
  k_normalize<<<dim3(Bb * Nn), dim3(Hh), 0, stream>>>(emb, z);

  size_t smem = (size_t)(16 * LDSLD + 32) * sizeof(float) + 16 * sizeof(int); // ~128.7 KB (LDS 320KB/WGP)
  k_sim_topk<<<dim3(Bb * NT), dim3(256), smem, stream>>>(z, topk, Aout, nidx, nw);

  k_gather<<<dim3(Bb * Nn), dim3(Hh), 0, stream>>>(emb, nidx, nw, mbuf);
  k_gemm_relu<<<gTiles, b128, 0, stream>>>(mbuf, Wg0T, bg0, h1);
  k_gather<<<dim3(Bb * Nn), dim3(Hh), 0, stream>>>(h1, nidx, nw, mbuf);
  k_gemm_relu<<<gTiles, b128, 0, stream>>>(mbuf, Wg1T, bg1, h2b);
  k_head<<<dim3(Bb * Nn), dim3(Hh / 2), 0, stream>>>(h2b, Wh1, bh1, Wh2, bh2, logits);
}
